// MultiHeadAttention_84567906058366
// MI455X (gfx1250) — matrix operand order repaired and checked by reading, not compiled, not run on hardware
//
#include <hip/hip_runtime.h>

// ---------------------------------------------------------------- types ----
typedef __bf16 bf16;
typedef __attribute__((ext_vector_type(16))) __bf16 bf16x16;
typedef __attribute__((ext_vector_type(8)))  float  f32x8;
typedef __attribute__((ext_vector_type(4)))  float  f32x4;
typedef __attribute__((ext_vector_type(4)))  unsigned int u32x4;
typedef __attribute__((ext_vector_type(2)))  unsigned int u32x2;
typedef __attribute__((ext_vector_type(4)))  int i32x4;

#define B_   4
#define S_   2048
#define D_   1024
#define H_   16
#define HD_  64
#define M1_  (B_ * S_)      // 8192
#define N1_  (3 * D_)       // 3072

// gfx1250 async global->LDS copy (ASYNCcnt-tracked), guarded so a toolchain
// without the builtin falls back to the synchronous VGPR-staged copy.
#if defined(__has_builtin)
#  if __has_builtin(__builtin_amdgcn_global_load_async_to_lds_b128)
#    define HAVE_ASYNC_LDS 1
#  endif
#endif
#ifndef HAVE_ASYNC_LDS
#  define HAVE_ASYNC_LDS 0
#endif

#if HAVE_ASYNC_LDS
__device__ inline void async_cp16(const bf16* gsrc, bf16* ldst) {
    // param 1: int4 in device/global address space (from compiler diagnostic);
    // param 2: LDS destination; params 3/4: imm offset, cache policy.
    __builtin_amdgcn_global_load_async_to_lds_b128(
        (__attribute__((address_space(1))) i32x4*)(gsrc),
        (__attribute__((address_space(3))) i32x4*)(ldst),
        /*offset=*/0, /*cpol=*/0);
}
#endif

union FragU { bf16x16 v; u32x4 u[2]; };

__device__ inline bf16x16 load_frag(const bf16* p0, const bf16* p1) {
    FragU f;
    f.u[0] = *(const u32x4*)p0;
    f.u[1] = *(const u32x4*)p1;
    return f.v;
}

__device__ inline f32x8 wmma_bf16(bf16x16 a, bf16x16 b, f32x8 c) {
    return __builtin_amdgcn_wmma_f32_16x16x32_bf16(
        /*neg_a=*/false, a, /*neg_b=*/false, b,
        /*c_mod=*/(short)0, c, /*reuse_a=*/false, /*reuse_b=*/false);
}

__device__ inline f32x8 zero8() {
    f32x8 z = {0.f, 0.f, 0.f, 0.f, 0.f, 0.f, 0.f, 0.f};
    return z;
}

// ------------------------------------------------ GEMM1: qkv = x @ Wqkv ----
// X [8192,1024] f32, W [1024,3072] f32. Epilogue scatters to:
//   Q [B,H,S,Hd] bf16 (pre-scaled by 1/8), K [B,H,S,Hd] bf16, V [B,H,Hd,S] bf16.
__global__ __launch_bounds__(256) void k_gemm_qkv(
    const float* __restrict__ X, const float* __restrict__ W,
    const float* __restrict__ bias,
    bf16* __restrict__ Qb, bf16* __restrict__ Kb, bf16* __restrict__ Vt)
{
    __shared__ __attribute__((aligned(16))) bf16 As[128][32];  // [m][k]
    __shared__ __attribute__((aligned(16))) bf16 Bs[128][32];  // [n][k]

    const int tid  = threadIdx.x;
    const int lane = tid & 31;
    const int wid  = tid >> 5;
    const int l16  = lane & 15, lh = lane >> 4;
    const int mbase = blockIdx.y * 128;
    const int nbase = blockIdx.x * 128;
    const int wm = (wid & 3) * 32;   // wave row offset in tile
    const int wn = (wid >> 2) * 64;  // wave col offset in tile

    f32x8 acc[2][4];
#pragma unroll
    for (int i = 0; i < 2; ++i)
#pragma unroll
        for (int j = 0; j < 4; ++j) acc[i][j] = zero8();

    // register double-buffer: global loads for tile k+1 overlap tile-k WMMAs
    f32x4 ra[4], rb[4];
#pragma unroll
    for (int i = 0; i < 4; ++i) {
        int id = tid + i * 256;
        int r = id >> 3, c = id & 7;
        ra[i] = *(const f32x4*)(X + (size_t)(mbase + r) * 1024 + c * 4);
        int kr = id >> 5, nc = id & 31;
        rb[i] = *(const f32x4*)(W + (size_t)kr * N1_ + nbase + nc * 4);
    }

    for (int kt = 0; kt < 1024; kt += 32) {
        __syncthreads();
        // commit prefetched tile to LDS (f32 -> bf16)
#pragma unroll
        for (int i = 0; i < 4; ++i) {
            int id = tid + i * 256;
            int r = id >> 3, c = id & 7;
            union { bf16 h[4]; u32x2 u; } p;
            p.h[0] = (bf16)ra[i][0]; p.h[1] = (bf16)ra[i][1];
            p.h[2] = (bf16)ra[i][2]; p.h[3] = (bf16)ra[i][3];
            *(u32x2*)&As[r][c * 4] = p.u;
            int kr = id >> 5, nc = id & 31;
#pragma unroll
            for (int j = 0; j < 4; ++j) Bs[nc * 4 + j][kr] = (bf16)rb[i][j];
        }
        __syncthreads();

        // prefetch next tile while the WMMAs below execute
        if (kt + 32 < 1024) {
#pragma unroll
            for (int i = 0; i < 4; ++i) {
                int id = tid + i * 256;
                int r = id >> 3, c = id & 7;
                ra[i] = *(const f32x4*)(X + (size_t)(mbase + r) * 1024 + (kt + 32) + c * 4);
                int kr = id >> 5, nc = id & 31;
                rb[i] = *(const f32x4*)(W + (size_t)(kt + 32 + kr) * N1_ + nbase + nc * 4);
            }
        }

        bf16x16 af[2];
#pragma unroll
        for (int mi = 0; mi < 2; ++mi) {
            const bf16* p = &As[wm + mi * 16 + l16][lh * 8];
            af[mi] = load_frag(p, p + 16);
        }
#pragma unroll
        for (int ni = 0; ni < 4; ++ni) {
            const bf16* p = &Bs[wn + ni * 16 + l16][lh * 8];
            bf16x16 bfr = load_frag(p, p + 16);
#pragma unroll
            for (int mi = 0; mi < 2; ++mi)
                acc[mi][ni] = wmma_bf16(af[mi], bfr, acc[mi][ni]);
        }
    }

    // epilogue: scatter to Q/K/V
#pragma unroll
    for (int mi = 0; mi < 2; ++mi)
#pragma unroll
        for (int ni = 0; ni < 4; ++ni)
#pragma unroll
            for (int r = 0; r < 8; ++r) {
                int row = mbase + wm + mi * 16 + lh * 8 + r;   // 0..8191
                int col = nbase + wn + ni * 16 + l16;          // 0..3071
                float v = acc[mi][ni][r] + bias[col];
                int which = col >> 10;
                int d  = col & 1023;
                int h  = d >> 6;
                int hd = d & 63;
                int b  = row >> 11;
                int s  = row & 2047;
                int bh = b * H_ + h;
                if (which == 0) {
                    Qb[((size_t)bh * S_ + s) * HD_ + hd] = (bf16)(v * 0.125f);
                } else if (which == 1) {
                    Kb[((size_t)bh * S_ + s) * HD_ + hd] = (bf16)v;
                } else {
                    Vt[((size_t)bh * HD_ + hd) * S_ + s] = (bf16)v;
                }
            }
}

// ----------------------------------------------- flash attention (causal) ---
// grid: (qt=S/128, bh=B*H). 8 waves, each owns 16 query rows.
// K/V tiles double-buffered; async global->LDS when available.
__global__ __launch_bounds__(256) void k_attn(
    const bf16* __restrict__ Qb, const bf16* __restrict__ Kb,
    const bf16* __restrict__ Vt, bf16* __restrict__ At)
{
    __shared__ __attribute__((aligned(16))) bf16 Ks[2][64][64];  // [buf][key][hd]
    __shared__ __attribute__((aligned(16))) bf16 Vs[2][64][64];  // [buf][hd][key]
    __shared__ __attribute__((aligned(16))) bf16 Ps[8][16][64];  // wave-private P

    const int tid  = threadIdx.x;
    const int lane = tid & 31;
    const int wid  = tid >> 5;
    const int l16  = lane & 15, lh = lane >> 4;
    const int bh   = blockIdx.y;
    const int qt   = blockIdx.x;
    const int qbase = qt * 128;
    const int q0    = qbase + wid * 16;

    // Q fragments (registers for whole kernel); scale 1/sqrt(Hd) pre-folded
    const bf16* qp = Qb + ((size_t)bh * S_ + q0 + l16) * HD_;
    bf16x16 qf[2];
#pragma unroll
    for (int s = 0; s < 2; ++s) {
        int base = s * 32 + lh * 8;
        qf[s] = load_frag(qp + base, qp + base + 16);
    }

    f32x8 of[4];
#pragma unroll
    for (int j = 0; j < 4; ++j) of[j] = zero8();
    float mrow[8], lrow[8];
#pragma unroll
    for (int r = 0; r < 8; ++r) { mrow[r] = -1.0e30f; lrow[r] = 0.0f; }

    const int nkt = 2 * qt + 2;  // causal: keys up to qbase+127

#if HAVE_ASYNC_LDS
    // issue tile 0 into buffer 0 (4 async b128 per wave per tile)
    {
        const bf16* kg = Kb + ((size_t)bh * S_) * HD_;
#pragma unroll
        for (int i = 0; i < 2; ++i) {
            int idx = tid + i * 256;
            async_cp16(kg + idx * 8, &Ks[0][0][0] + idx * 8);
        }
#pragma unroll
        for (int i = 0; i < 2; ++i) {
            int idx = tid + i * 256;
            int hd = idx >> 3, c = idx & 7;
            async_cp16(Vt + ((size_t)bh * HD_ + hd) * S_ + c * 8, &Vs[0][hd][c * 8]);
        }
    }
#endif

    for (int kt = 0; kt < nkt; ++kt) {
        const int kbase = kt * 64;
        const int cur = kt & 1;
        __syncthreads();   // prior iteration finished reading buf (cur^1)
#if HAVE_ASYNC_LDS
        if (kt + 1 < nkt) {
            const int nb = cur ^ 1;
            const int kb2 = (kt + 1) * 64;
            const bf16* kg = Kb + ((size_t)bh * S_ + kb2) * HD_;
#pragma unroll
            for (int i = 0; i < 2; ++i) {
                int idx = tid + i * 256;
                async_cp16(kg + idx * 8, &Ks[nb][0][0] + idx * 8);
            }
#pragma unroll
            for (int i = 0; i < 2; ++i) {
                int idx = tid + i * 256;
                int hd = idx >> 3, c = idx & 7;
                async_cp16(Vt + ((size_t)bh * HD_ + hd) * S_ + kb2 + c * 8,
                           &Vs[nb][hd][c * 8]);
            }
            asm volatile("s_wait_asynccnt 0x4" ::: "memory");  // drain current buf
        } else {
            asm volatile("s_wait_asynccnt 0x0" ::: "memory");
        }
#else
        // synchronous fallback copy into buffer `cur`
        {
            const u32x4* kg = (const u32x4*)(Kb + ((size_t)bh * S_ + kbase) * HD_);
            u32x4* ks = (u32x4*)&Ks[cur][0][0];
            ks[tid]       = kg[tid];
            ks[tid + 256] = kg[tid + 256];
        }
#pragma unroll
        for (int i = 0; i < 2; ++i) {
            int id = tid + i * 256;
            int hd = id >> 3, c = id & 7;
            *(u32x4*)&Vs[cur][hd][c * 8] =
                *(const u32x4*)(Vt + ((size_t)bh * HD_ + hd) * S_ + kbase + c * 8);
        }
#endif
        __syncthreads();

        // S = Q @ K^T  (16 queries x 64 keys per wave)
        f32x8 sf[4];
#pragma unroll
        for (int j = 0; j < 4; ++j) sf[j] = zero8();
#pragma unroll
        for (int ks = 0; ks < 2; ++ks)
#pragma unroll
            for (int j = 0; j < 4; ++j) {
                const bf16* p = &Ks[cur][j * 16 + l16][ks * 32 + lh * 8];
                bf16x16 bfr = load_frag(p, p + 16);
                sf[j] = wmma_bf16(qf[ks], bfr, sf[j]);
            }

        // causal mask + online softmax
        const int qrow = qbase + wid * 16 + lh * 8;
#pragma unroll
        for (int r = 0; r < 8; ++r) {
            const int qg = qrow + r;
            float mx = -1.0e30f;
#pragma unroll
            for (int j = 0; j < 4; ++j) {
                int kg = kbase + j * 16 + l16;
                float sv = sf[j][r];
                sv = (kg > qg) ? -1.0e30f : sv;
                sf[j][r] = sv;
                mx = fmaxf(mx, sv);
            }
#pragma unroll
            for (int off = 1; off < 16; off <<= 1)
                mx = fmaxf(mx, __shfl_xor(mx, off, 32));
            const float mnew = fmaxf(mrow[r], mx);
            const float corr = __expf(mrow[r] - mnew);
            mrow[r] = mnew;
            float rs = 0.0f;
#pragma unroll
            for (int j = 0; j < 4; ++j) {
                float e = __expf(sf[j][r] - mnew);
                Ps[wid][lh * 8 + r][j * 16 + l16] = (bf16)e;
                rs += e;
            }
#pragma unroll
            for (int off = 1; off < 16; off <<= 1)
                rs += __shfl_xor(rs, off, 32);
            lrow[r] = lrow[r] * corr + rs;
#pragma unroll
            for (int j = 0; j < 4; ++j) of[j][r] *= corr;
        }
        // drain LDS writes before re-reading P in A-fragment layout (same wave)
        asm volatile("s_wait_dscnt 0x0" ::: "memory");

        // O += P @ V
#pragma unroll
        for (int ks = 0; ks < 2; ++ks) {
            const bf16* pp = &Ps[wid][l16][ks * 32 + lh * 8];
            bf16x16 pf = load_frag(pp, pp + 16);
#pragma unroll
            for (int j = 0; j < 4; ++j) {
                const bf16* p = &Vs[cur][j * 16 + l16][ks * 32 + lh * 8];
                bf16x16 bfr = load_frag(p, p + 16);
                of[j] = wmma_bf16(pf, bfr, of[j]);
            }
        }
    }

    // normalize + store to At [B*S, D] bf16 (col = h*64+hd)
    const int b = bh >> 4, h = bh & 15;
#pragma unroll
    for (int r = 0; r < 8; ++r) {
        const float inv = 1.0f / lrow[r];
        const size_t rowg = (size_t)b * S_ + qbase + wid * 16 + lh * 8 + r;
#pragma unroll
        for (int j = 0; j < 4; ++j)
            At[rowg * D_ + h * 64 + j * 16 + l16] = (bf16)(of[j][r] * inv);
    }
}

// -------------------------------------- GEMM2: out = attn @ W_out + b_out ---
__global__ __launch_bounds__(256) void k_gemm_out(
    const bf16* __restrict__ A, const float* __restrict__ W,
    const float* __restrict__ bias, float* __restrict__ Out)
{
    __shared__ __attribute__((aligned(16))) bf16 As[128][32];
    __shared__ __attribute__((aligned(16))) bf16 Bs[128][32];

    const int tid  = threadIdx.x;
    const int lane = tid & 31;
    const int wid  = tid >> 5;
    const int l16  = lane & 15, lh = lane >> 4;
    const int mbase = blockIdx.y * 128;
    const int nbase = blockIdx.x * 128;
    const int wm = (wid & 3) * 32;
    const int wn = (wid >> 2) * 64;

    f32x8 acc[2][4];
#pragma unroll
    for (int i = 0; i < 2; ++i)
#pragma unroll
        for (int j = 0; j < 4; ++j) acc[i][j] = zero8();

    // register double-buffer
    u32x4 ra[2];
    f32x4 rb[4];
#pragma unroll
    for (int i = 0; i < 2; ++i) {
        int id = tid + i * 256;
        int r = id >> 2, c = id & 3;
        ra[i] = *(const u32x4*)(A + (size_t)(mbase + r) * 1024 + c * 8);
    }
#pragma unroll
    for (int i = 0; i < 4; ++i) {
        int id = tid + i * 256;
        int kr = id >> 5, nc = id & 31;
        rb[i] = *(const f32x4*)(W + (size_t)kr * 1024 + nbase + nc * 4);
    }

    for (int kt = 0; kt < 1024; kt += 32) {
        __syncthreads();
#pragma unroll
        for (int i = 0; i < 2; ++i) {
            int id = tid + i * 256;
            int r = id >> 2, c = id & 3;
            *(u32x4*)&As[r][c * 8] = ra[i];
        }
#pragma unroll
        for (int i = 0; i < 4; ++i) {
            int id = tid + i * 256;
            int kr = id >> 5, nc = id & 31;
#pragma unroll
            for (int j = 0; j < 4; ++j) Bs[nc * 4 + j][kr] = (bf16)rb[i][j];
        }
        __syncthreads();

        if (kt + 32 < 1024) {
#pragma unroll
            for (int i = 0; i < 2; ++i) {
                int id = tid + i * 256;
                int r = id >> 2, c = id & 3;
                ra[i] = *(const u32x4*)(A + (size_t)(mbase + r) * 1024 + (kt + 32) + c * 8);
            }
#pragma unroll
            for (int i = 0; i < 4; ++i) {
                int id = tid + i * 256;
                int kr = id >> 5, nc = id & 31;
                rb[i] = *(const f32x4*)(W + (size_t)(kt + 32 + kr) * 1024 + nbase + nc * 4);
            }
        }

        bf16x16 af[2];
#pragma unroll
        for (int mi = 0; mi < 2; ++mi) {
            const bf16* p = &As[wm + mi * 16 + l16][lh * 8];
            af[mi] = load_frag(p, p + 16);
        }
#pragma unroll
        for (int ni = 0; ni < 4; ++ni) {
            const bf16* p = &Bs[wn + ni * 16 + l16][lh * 8];
            bf16x16 bfr = load_frag(p, p + 16);
#pragma unroll
            for (int mi = 0; mi < 2; ++mi)
                acc[mi][ni] = wmma_bf16(af[mi], bfr, acc[mi][ni]);
        }
    }

#pragma unroll
    for (int mi = 0; mi < 2; ++mi)
#pragma unroll
        for (int ni = 0; ni < 4; ++ni)
#pragma unroll
            for (int r = 0; r < 8; ++r) {
                int row = mbase + wm + mi * 16 + lh * 8 + r;
                int col = nbase + wn + ni * 16 + l16;
                Out[(size_t)row * 1024 + col] = acc[mi][ni][r] + bias[col];
            }
}

// ------------------------------------------------------------------ host ----
extern "C" void kernel_launch(void* const* d_in, const int* in_sizes, int n_in,
                              void* d_out, int out_size, void* d_ws, size_t ws_size,
                              hipStream_t stream) {
    const float* x     = (const float*)d_in[0];
    const float* W_qkv = (const float*)d_in[1];
    const float* b_qkv = (const float*)d_in[2];
    const float* W_out = (const float*)d_in[3];
    const float* b_out = (const float*)d_in[4];
    float* out = (float*)d_out;

    // workspace: Q 16MB | K 16MB | V^T 16MB | attn 16MB  (all bf16)
    char* ws = (char*)d_ws;
    bf16* Qb = (bf16*)(ws);
    bf16* Kb = (bf16*)(ws + ((size_t)16 << 20));
    bf16* Vt = (bf16*)(ws + ((size_t)32 << 20));
    bf16* At = (bf16*)(ws + ((size_t)48 << 20));

    // 1) QKV projection: [8192,1024] x [1024,3072]
    k_gemm_qkv<<<dim3(N1_ / 128, M1_ / 128), 256, 0, stream>>>(
        x, W_qkv, b_qkv, Qb, Kb, Vt);

    // 2) causal flash attention per (b,h)
    k_attn<<<dim3(S_ / 128, B_ * H_), 256, 0, stream>>>(Qb, Kb, Vt, At);

    // 3) output projection: [8192,1024] x [1024,1024]
    k_gemm_out<<<dim3(D_ / 128, M1_ / 128), 256, 0, stream>>>(
        At, W_out, b_out, out);
}
